// T5Attention_36438502539938
// MI455X (gfx1250) — compile-verified, matrix operand order checked by reading, not run on hardware
//
#include <hip/hip_runtime.h>
#include <hip/hip_bf16.h>
#include <math.h>

// ---------------------------------------------------------------------------
// T5 attention for MI455X / gfx1250 (wave32, WMMA 16x16x32 f16->f32)
// B=2, S=2048, D=1024, H=16, HD=64, NUM_BUCKETS=32, MAX_DISTANCE=128
// Outputs concat: attn_output [B,S,D] | attn_weights [B,H,S,S] | pos_bias [1,H,S,S]
//
// Pipeline:
//   K0 bias_kernel   : pos_bias region (elementwise, streaming)
//   K1 proj_kernel x3: Q,K ([B,H,S,HD] f16), V transposed ([B,H,HD,S] f16)
//   K2 attn_fused    : per 16-row q-block: scores(WMMA) + inline bias + mask
//                      -> LDS (16x2048 f32), softmax in LDS, stream weights
//                      out once, then weights@V with A-fragments from LDS.
//   K3 oproj_kernel  : output projection (WMMA)
// ---------------------------------------------------------------------------

#define Bc 2
#define Sc 2048
#define Dc 1024
#define Hc 16
#define HDc 64
#define SPAD 2068                 // padded LDS row stride (floats); 2068%64=20

typedef _Float16 v16h __attribute__((ext_vector_type(16)));
typedef _Float16 v8h  __attribute__((ext_vector_type(8)));
typedef float    v8f  __attribute__((ext_vector_type(8)));

__device__ __forceinline__ v8f wmma32(v16h a, v16h b, v8f c) {
  return __builtin_amdgcn_wmma_f32_16x16x32_f16(false, a, false, b, (short)0, c,
                                                false, false);
}

// A/B fragment from an f16 row-major source. `row` = this lane's row
// (row index = lane&15), k window [k0,k0+32). Lanes 0-15 hold K
// {k0+0..7, k0+16..23}; lanes 16-31 hold K {k0+8..15, k0+24..31}.
__device__ __forceinline__ v16h frag_f16(const _Float16* __restrict__ row, int k0) {
  const int lane = threadIdx.x & 31;
  const _Float16* p = row + k0 + ((lane & 16) ? 8 : 0);
  v8h lo = *(const v8h*)(p);
  v8h hi = *(const v8h*)(p + 16);
  v16h r;
#pragma unroll
  for (int i = 0; i < 8; ++i) { r[i] = lo[i]; r[i + 8] = hi[i]; }
  return r;
}

// Same fragment, converting from an f32 source (works for global or LDS ptrs).
__device__ __forceinline__ v16h frag_f32(const float* __restrict__ row, int k0) {
  const int lane = threadIdx.x & 31;
  const float* p = row + k0 + ((lane & 16) ? 8 : 0);
  const float4* p4 = (const float4*)p;
  float4 x0 = p4[0];
  float4 x1 = p4[1];
  float4 x2 = p4[4];   // p + 16 floats
  float4 x3 = p4[5];
  v16h r;
  r[0]  = (_Float16)x0.x; r[1]  = (_Float16)x0.y; r[2]  = (_Float16)x0.z; r[3]  = (_Float16)x0.w;
  r[4]  = (_Float16)x1.x; r[5]  = (_Float16)x1.y; r[6]  = (_Float16)x1.z; r[7]  = (_Float16)x1.w;
  r[8]  = (_Float16)x2.x; r[9]  = (_Float16)x2.y; r[10] = (_Float16)x2.z; r[11] = (_Float16)x2.w;
  r[12] = (_Float16)x3.x; r[13] = (_Float16)x3.y; r[14] = (_Float16)x3.z; r[15] = (_Float16)x3.w;
  return r;
}

// T5 relative-position bucket (bidirectional, 32 buckets, max_distance 128).
__device__ __forceinline__ int t5_bucket(int rp) {
  int bucket = (rp > 0) ? 16 : 0;
  int a = (rp < 0) ? -rp : rp;
  int v;
  if (a < 8) {
    v = a;
  } else {
    v = 8 + (int)(__logf((float)a * 0.125f) * (8.0f / __logf(16.0f)));
    v = (v > 15) ? 15 : v;
  }
  return bucket + v;
}

// ---------------------------------------------------------------------------
// K0: position bias  pos_bias[h][q][k] = rel_bias[bucket(k-q)][h]
// ---------------------------------------------------------------------------
__global__ void t5_bias_kernel(const float* __restrict__ rel_bias,
                               float* __restrict__ pos_bias) {
  long idx = (long)blockIdx.x * blockDim.x + threadIdx.x;   // H*S*S
  int k = (int)(idx & (Sc - 1));
  int q = (int)((idx >> 11) & (Sc - 1));
  int h = (int)(idx >> 22);
  pos_bias[idx] = rel_bias[t5_bucket(k - q) * Hc + h];
}

// ---------------------------------------------------------------------------
// K1: projection GEMM  out[m,n] = sum_d X[m,d] * W[n,d]   (M=4096,N=1024,K=1024)
// mode 0/1: store as [B,H,S,HD] f16   (Q, K)
// mode 2  : store as [B,H,HD,S] f16   (V transposed for the AV GEMM)
// ---------------------------------------------------------------------------
__global__ void t5_proj_kernel(const float* __restrict__ X,
                               const float* __restrict__ W,
                               _Float16* __restrict__ out, int mode) {
  const int wave  = threadIdx.x >> 5;
  const int tile  = blockIdx.x * (blockDim.x >> 5) + wave;  // 256*64 tiles
  const int ntile = tile & 63;
  const int mtile = tile >> 6;
  const int lane  = threadIdx.x & 31;
  const int r     = lane & 15;

  const float* arow = X + (size_t)(mtile * 16 + r) * Dc;
  const float* brow = W + (size_t)(ntile * 16 + r) * Dc;

  v8f c = {};
#pragma unroll 4
  for (int k0 = 0; k0 < Dc; k0 += 32) {
    __builtin_prefetch(arow + k0 + 128, 0, 0);
    __builtin_prefetch(brow + k0 + 128, 0, 0);
    c = wmma32(frag_f32(arow, k0), frag_f32(brow, k0), c);
  }

  const int n = ntile * 16 + (lane & 15);
  const int h = n >> 6;
  const int d = n & (HDc - 1);
#pragma unroll
  for (int i = 0; i < 8; ++i) {
    int m  = mtile * 16 + i + ((lane >> 4) << 3);
    int bb = m >> 11;
    int s  = m & (Sc - 1);
    _Float16 val = (_Float16)c[i];
    if (mode == 2) {
      out[(((size_t)bb * Hc + h) * HDc + d) * Sc + s] = val;  // Vt [B,H,HD,S]
    } else {
      out[(((size_t)bb * Hc + h) * Sc + s) * HDc + d] = val;  // [B,H,S,HD]
    }
  }
}

// ---------------------------------------------------------------------------
// K2: fused attention per (b, h, 16-row q-block).
// 8 waves / 256 threads. LDS: scores [16][SPAD] f32 + partials [8][256] f32.
// ---------------------------------------------------------------------------
__global__ void t5_attn_fused_kernel(const _Float16* __restrict__ Q,
                                     const _Float16* __restrict__ Kf,
                                     const _Float16* __restrict__ Vt,
                                     const float* __restrict__ rel_bias,
                                     const int* __restrict__ mask,
                                     float* __restrict__ weights_out,
                                     float* __restrict__ att) {
  extern __shared__ float smem[];
  float* sc   = smem;                 // [16][SPAD] score block
  float* part = smem + 16 * SPAD;     // [8][256]  AV partial sums

  const int qtile = blockIdx.x;       // 0..127
  const int h     = blockIdx.y;
  const int b     = blockIdx.z;
  const long bh   = (long)b * Hc + h;
  const int wave  = threadIdx.x >> 5;
  const int lane  = threadIdx.x & 31;
  const int r     = lane & 15;

  // Q fragments for this 16-row block: loaded once, reused for all 128 ktiles.
  const _Float16* qrow = Q + ((bh * Sc + qtile * 16 + r) * HDc);
  const v16h qa0 = frag_f16(qrow, 0);
  const v16h qa1 = frag_f16(qrow, 32);

  // ---- Phase A: scores = Q K^T + bias, masked -> LDS -------------------
  for (int kt = wave; kt < 128; kt += 8) {
    const _Float16* krow = Kf + ((bh * Sc + kt * 16 + r) * HDc);
    v8f c = {};
    c = wmma32(qa0, frag_f16(krow, 0),  c);
    c = wmma32(qa1, frag_f16(krow, 32), c);

    const int kcol = kt * 16 + (lane & 15);
    const int mval = mask[b * Sc + kcol];
#pragma unroll
    for (int i = 0; i < 8; ++i) {
      int m = i + ((lane >> 4) << 3);
      int q = qtile * 16 + m;
      float s = c[i] + rel_bias[t5_bucket(kcol - q) * Hc + h];
      if (mval == 0) s = -1e9f;
      sc[m * SPAD + kcol] = s;
    }
  }
  __syncthreads();

  // ---- Phase B: softmax over each of the 16 rows (in LDS) --------------
  {
    const int row = threadIdx.x >> 4;    // 16 threads per row
    const int jj  = threadIdx.x & 15;
    float* prow = sc + row * SPAD;

    float m = -3.402823466e38f;
    for (int j = jj; j < Sc; j += 16) m = fmaxf(m, prow[j]);
#pragma unroll
    for (int off = 1; off < 16; off <<= 1) m = fmaxf(m, __shfl_xor(m, off, 32));

    float ssum = 0.0f;
    for (int j = jj; j < Sc; j += 16) {
      float e = __expf(prow[j] - m);
      prow[j] = e;
      ssum += e;
    }
#pragma unroll
    for (int off = 1; off < 16; off <<= 1) ssum += __shfl_xor(ssum, off, 32);

    const float inv = 1.0f / ssum;
    for (int j = jj; j < Sc; j += 16) prow[j] *= inv;
  }
  __syncthreads();

  // ---- Phase C: stream normalized weights to HBM (coalesced b128) ------
  {
    float* wbase = weights_out + bh * (long)Sc * Sc + (long)(qtile * 16) * Sc;
    for (int i = threadIdx.x; i < 16 * (Sc / 4); i += 256) {
      int row = i >> 9;           // Sc/4 = 512 float4 per row
      int c4  = i & 511;
      float4 v = *(const float4*)(sc + row * SPAD + c4 * 4);
      *(float4*)(wbase + (long)row * Sc + c4 * 4) = v;
    }
  }

  // ---- Phase D: att_block = weights @ V  (A from LDS, B from Vt) -------
  {
    const int ntile = wave & 3;       // N = 64 -> 4 ntiles
    const int khalf = wave >> 2;      // split K=2048 across 2 waves
    const _Float16* brow = Vt + ((bh * HDc + ntile * 16 + r) * (long)Sc);
    const float* arow = sc + r * SPAD;

    v8f c = {};
    const int kbeg = khalf * (Sc / 2);
#pragma unroll 4
    for (int k0 = kbeg; k0 < kbeg + Sc / 2; k0 += 32) {
      c = wmma32(frag_f32(arow, k0), frag_f16(brow, k0), c);
    }
#pragma unroll
    for (int i = 0; i < 8; ++i) part[wave * 256 + lane * 8 + i] = c[i];
    __syncthreads();

    if (khalf == 0) {
#pragma unroll
      for (int i = 0; i < 8; ++i) c[i] += part[(wave + 4) * 256 + lane * 8 + i];
      const int n = ntile * 16 + (lane & 15);
#pragma unroll
      for (int i = 0; i < 8; ++i) {
        int m = qtile * 16 + i + ((lane >> 4) << 3);
        att[((long)b * Sc + m) * Dc + h * HDc + n] = c[i];
      }
    }
  }
}

// ---------------------------------------------------------------------------
// K3: attn_output = att @ Wo^T   (M=4096, N=1024, K=1024), f32 out.
// ---------------------------------------------------------------------------
__global__ void t5_oproj_kernel(const float* __restrict__ att,
                                const float* __restrict__ Wo,
                                float* __restrict__ out) {
  const int wave  = threadIdx.x >> 5;
  const int tile  = blockIdx.x * (blockDim.x >> 5) + wave;  // 256*64
  const int ntile = tile & 63;
  const int mtile = tile >> 6;
  const int lane  = threadIdx.x & 31;
  const int r     = lane & 15;

  const float* arow = att + (size_t)(mtile * 16 + r) * Dc;
  const float* brow = Wo  + (size_t)(ntile * 16 + r) * Dc;

  v8f c = {};
#pragma unroll 4
  for (int k0 = 0; k0 < Dc; k0 += 32) {
    __builtin_prefetch(arow + k0 + 128, 0, 0);
    __builtin_prefetch(brow + k0 + 128, 0, 0);
    c = wmma32(frag_f32(arow, k0), frag_f32(brow, k0), c);
  }

  const int n = ntile * 16 + (lane & 15);
#pragma unroll
  for (int i = 0; i < 8; ++i) {
    int m = mtile * 16 + i + ((lane >> 4) << 3);
    out[(long)m * Dc + n] = c[i];
  }
}

// ---------------------------------------------------------------------------
// Launch
// ---------------------------------------------------------------------------
extern "C" void kernel_launch(void* const* d_in, const int* in_sizes, int n_in,
                              void* d_out, int out_size, void* d_ws, size_t ws_size,
                              hipStream_t stream) {
  const float* hidden   = (const float*)d_in[0];   // [B,S,D]
  const int*   mask     = (const int*)  d_in[1];   // [B,1,1,S]
  const float* Wq       = (const float*)d_in[2];   // [D,D]
  const float* Wk       = (const float*)d_in[3];
  const float* Wv       = (const float*)d_in[4];
  const float* Wo       = (const float*)d_in[5];
  const float* rel_bias = (const float*)d_in[6];   // [32,16]

  float* out      = (float*)d_out;
  float* out_attn = out;                                        //  4,194,304
  float* out_w    = out_attn + (size_t)Bc * Sc * Dc;            // 134,217,728
  float* out_bias = out_w + (size_t)Bc * Hc * Sc * Sc;          //  67,108,864

  // Workspace: Q | K | Vt (f16, 8 MB each) | att (f32, 16 MB) => 40 MB
  _Float16* Qb = (_Float16*)d_ws;
  _Float16* Kb = Qb + (size_t)Bc * Hc * Sc * HDc;
  _Float16* Vt = Kb + (size_t)Bc * Hc * Sc * HDc;
  float*    att = (float*)(Vt + (size_t)Bc * Hc * Sc * HDc);

  // K0: position bias (H*S*S elements)
  t5_bias_kernel<<<(Hc * Sc * Sc) / 256, 256, 0, stream>>>(rel_bias, out_bias);

  // K1: Q/K/V projections (16384 tiles, 8 waves/block)
  t5_proj_kernel<<<2048, 256, 0, stream>>>(hidden, Wq, Qb, 0);
  t5_proj_kernel<<<2048, 256, 0, stream>>>(hidden, Wk, Kb, 1);
  t5_proj_kernel<<<2048, 256, 0, stream>>>(hidden, Wv, Vt, 2);

  // K2: fused scores+softmax+AV; LDS = 16*SPAD + 8*256 floats (~137 KB)
  const size_t smem_bytes = (size_t)(16 * SPAD + 8 * 256) * sizeof(float);
  t5_attn_fused_kernel<<<dim3(Sc / 16, Hc, Bc), 256, smem_bytes, stream>>>(
      Qb, Kb, Vt, rel_bias, mask, out_w, att);

  // K3: output projection
  t5_oproj_kernel<<<2048, 256, 0, stream>>>(att, Wo, out_attn);
}
